// AttnBlock_56616258895923
// MI455X (gfx1250) — matrix operand order repaired and checked by reading, not compiled, not run on hardware
//
#include <hip/hip_runtime.h>
#include <math.h>

#define B_ 8
#define C_ 256
#define N_ 2048

typedef __attribute__((ext_vector_type(16))) _Float16 v16h;
typedef __attribute__((ext_vector_type(8)))  float    v8f;
typedef _Float16 f16;

__device__ __forceinline__ v8f wmma_f32(v16h a, v16h b, v8f c) {
  // 8 args: (neg_a, A, neg_b, B, c_mod, C, reuse_a, reuse_b)
  return __builtin_amdgcn_wmma_f32_16x16x32_f16(false, a, false, b, (short)0, c,
                                                false, false);
}

// A fragment: 16x32 (MxK), row-major source with leading dim lda.
// lane<16: M=lane, elems 0..15 -> K 0..15
// lane>=16: M=lane-16, elems 0..15 -> K 16..31
__device__ __forceinline__ v16h load_A_frag(const f16* __restrict__ base, int lda) {
  int lane = threadIdx.x & 31;
  const f16* p = base + (size_t)(lane & 15) * lda + ((lane >> 4) << 4);
  v16h a;
#pragma unroll
  for (int j = 0; j < 8; ++j) { a[j] = p[j]; a[j + 8] = p[8 + j]; }
  return a;
}

// B fragment: 32x16 (KxN) from row-major B[k][n] with leading dim ldb.
// n = lane&15 ; lane<16: K=elem, lane>=16: K=16+elem
__device__ __forceinline__ v16h load_B_frag(const f16* __restrict__ base, int ldb) {
  int lane = threadIdx.x & 31;
  const f16* p = base + (size_t)((lane >> 4) << 4) * ldb + (lane & 15);
  v16h b;
#pragma unroll
  for (int j = 0; j < 16; ++j) b[j] = p[(size_t)j * ldb];
  return b;
}

// B fragment where the source is stored row-major as M[n][k] (transposed view):
// each lane reads 16 contiguous halves (32 bytes).
__device__ __forceinline__ v16h load_Bt_frag(const f16* __restrict__ base, int ld) {
  int lane = threadIdx.x & 31;
  const f16* p = base + (size_t)(lane & 15) * ld + ((lane >> 4) << 4);
  v16h b;
#pragma unroll
  for (int j = 0; j < 16; ++j) b[j] = p[j];
  return b;
}

// ---------------- BatchNorm statistics: fold into h = x*s + t ----------------
__global__ void bn_stats_k(const float* __restrict__ x, const float* __restrict__ gamma,
                           const float* __restrict__ beta, float* __restrict__ s,
                           float* __restrict__ t) {
  const int c = blockIdx.x;
  const int BN = B_ * N_;
  float sum = 0.f, sq = 0.f;
  for (int i = threadIdx.x; i < BN; i += 256) {
    int b = i >> 11, n = i & (N_ - 1);
    float v = x[(size_t)b * C_ * N_ + (size_t)c * N_ + n];
    sum += v; sq += v * v;
  }
  __shared__ float ssum[256], ssq[256];
  ssum[threadIdx.x] = sum; ssq[threadIdx.x] = sq;
  __syncthreads();
  for (int off = 128; off > 0; off >>= 1) {
    if ((int)threadIdx.x < off) {
      ssum[threadIdx.x] += ssum[threadIdx.x + off];
      ssq[threadIdx.x]  += ssq[threadIdx.x + off];
    }
    __syncthreads();
  }
  if (threadIdx.x == 0) {
    float mean = ssum[0] / BN;
    float var  = ssq[0] / BN - mean * mean;
    float rstd = rsqrtf(var + 1e-5f);
    float sc = rstd * gamma[c];
    s[c] = sc;
    t[c] = beta[c] - mean * sc;
  }
}

// ---------------- Normalize + convert to fp16, layout [B,C,N] ----------------
__global__ void bn_apply_k(const float* __restrict__ x, const float* __restrict__ s,
                           const float* __restrict__ t, f16* __restrict__ h) {
  size_t i = (size_t)blockIdx.x * 256 + threadIdx.x;
  int c = (int)((i >> 11) & (C_ - 1));
  h[i] = (f16)(x[i] * s[c] + t[c]);
}

// ---------------- Weights fp32 -> fp16 (Wq,Wk,Wv stacked; Wp separate) -------
__global__ void cvt_w_k(const float* __restrict__ Wq, const float* __restrict__ Wk,
                        const float* __restrict__ Wv, const float* __restrict__ Wp,
                        f16* __restrict__ W3, f16* __restrict__ Wp16) {
  int i = blockIdx.x * 256 + threadIdx.x;               // C_*C_ elems
  W3[i]                 = (f16)Wq[i];
  W3[C_ * C_ + i]       = (f16)Wk[i];
  W3[2 * C_ * C_ + i]   = (f16)Wv[i];
  Wp16[i]               = (f16)Wp[i];
}

// ---------------- Q/K/V projections: per-wave 32x32 WMMA tile ----------------
// Block = 8 waves covering full o range (8 x 32) for ONE n-tile, so the H strip
// is staged once in LDS and broadcast to all waves.
// out[mat][b][n][o] = (sum_c W[mat][o][c]*H[b][c][n] + bias[o]) * (mat==0 ? C^-0.5 : 1)
__global__ void proj_qkv_k(const f16* __restrict__ H, const f16* __restrict__ W3,
                           const float* __restrict__ bq, const float* __restrict__ bk,
                           const float* __restrict__ bv, f16* __restrict__ Q,
                           f16* __restrict__ K, f16* __restrict__ V) {
  __shared__ f16 Hsh[32][40];                           // 32(k) x 32(n), padded
  int wave = threadIdx.x >> 5, lane = threadIdx.x & 31, tid = threadIdx.x;
  int gwave = blockIdx.x * 8 + wave;                    // 12288 waves
  int tile = gwave & 511;                               // 8(o) x 64(n) tiles
  int mb   = gwave >> 9;                                // 0..23 (same within block)
  int mat  = mb >> 3, b = mb & 7;
  int oT = (tile & 7) * 32;                             // differs per wave
  int nT = (tile >> 3) * 32;                            // same within block
  const f16* A  = W3 + (size_t)mat * C_ * C_;           // [C,C] row-major
  const f16* Bm = H + (size_t)b * C_ * N_;              // [C,N] row-major
  v8f acc[2][2] = {};
  for (int k = 0; k < C_; k += 32) {
    {                                                   // cooperative H strip load
      int kk = tid >> 3, n4 = (tid & 7) << 2;
      const f16* hp = Bm + (size_t)(k + kk) * N_ + nT + n4;
#pragma unroll
      for (int j = 0; j < 4; ++j) Hsh[kk][n4 + j] = hp[j];
    }
    __syncthreads();
    v16h a0 = load_A_frag(A + (size_t)oT * C_ + k, C_);
    v16h a1 = load_A_frag(A + (size_t)(oT + 16) * C_ + k, C_);
    v16h b0 = load_B_frag(&Hsh[0][0], 40);
    v16h b1 = load_B_frag(&Hsh[0][16], 40);
    acc[0][0] = wmma_f32(a0, b0, acc[0][0]);
    acc[0][1] = wmma_f32(a0, b1, acc[0][1]);
    acc[1][0] = wmma_f32(a1, b0, acc[1][0]);
    acc[1][1] = wmma_f32(a1, b1, acc[1][1]);
    __syncthreads();
  }
  const float* bias = (mat == 0) ? bq : ((mat == 1) ? bk : bv);
  f16* out = ((mat == 0) ? Q : ((mat == 1) ? K : V)) + (size_t)b * N_ * C_;
  float scl = (mat == 0) ? 0.0625f : 1.0f;              // C^-0.5 folded into q
  int rowOff = (lane >> 4) << 3, colB = lane & 15;
#pragma unroll
  for (int mi = 0; mi < 2; ++mi)
#pragma unroll
    for (int ni = 0; ni < 2; ++ni)
#pragma unroll
      for (int r = 0; r < 8; ++r) {
        int o = oT + mi * 16 + r + rowOff;
        int n = nT + ni * 16 + colB;
        out[(size_t)n * C_ + o] = (f16)((acc[mi][ni][r] + bias[o]) * scl);
      }
}

// ---------------- Flash attention -------------------------------------------
// Block = 8 waves x 16 queries = 128 queries. K tile [32][C] and transposed V
// tile [C][32] are staged cooperatively in LDS once per 32-key step and shared
// by all waves (8x less L2 traffic than per-wave streaming).
__global__ void attn_k(const f16* __restrict__ Q, const f16* __restrict__ K,
                       const f16* __restrict__ V, f16* __restrict__ O) {
  __shared__ f16 Ksh[32][C_ + 8];                        // [key][c], padded
  __shared__ f16 Vsh[C_][40];                            // [c][key], padded
  __shared__ f16 Pl[8][16][32];                          // per-wave P staging
  int wave = threadIdx.x >> 5, lane = threadIdx.x & 31, tid = threadIdx.x;
  int b  = blockIdx.x >> 4;
  int gq = (blockIdx.x & 15) * 128 + wave * 16;          // query row base
  const f16* Qb = Q + (size_t)b * N_ * C_;
  const f16* Kb = K + (size_t)b * N_ * C_;
  const f16* Vb = V + (size_t)b * N_ * C_;
  int rowOff = (lane >> 4) << 3, colB = lane & 15;

  v16h qf[8];                                            // Q strip resident
#pragma unroll
  for (int kc = 0; kc < 8; ++kc)
    qf[kc] = load_A_frag(Qb + (size_t)gq * C_ + kc * 32, C_);

  v8f oacc[16] = {};                                     // O accum over C=256
  float mrow[8], lrow[8];
#pragma unroll
  for (int r = 0; r < 8; ++r) { mrow[r] = -1e30f; lrow[r] = 0.f; }

  for (int kt = 0; kt < N_; kt += 32) {                  // 32 keys per step
    // ---- cooperative stage of K tile and transposed V tile ----
#pragma unroll
    for (int it = 0; it < 2; ++it) {
      int s2 = tid + it * 256;                           // 512 slots of 16 halves
      int key = s2 >> 4, cg = (s2 & 15) << 4;
      const f16* kp = Kb + (size_t)(kt + key) * C_ + cg;
      const f16* vp = Vb + (size_t)(kt + key) * C_ + cg;
#pragma unroll
      for (int j = 0; j < 16; ++j) {
        Ksh[key][cg + j] = kp[j];
        Vsh[cg + j][key] = vp[j];
      }
    }
    if (kt + 32 < N_) {                                  // prefetch next tile
      int key = tid >> 4, cg = (tid & 15) << 4;
      __builtin_prefetch(Kb + (size_t)(kt + 32 + key) * C_ + cg, 0, 3);
      __builtin_prefetch(Vb + (size_t)(kt + 32 + key) * C_ + cg, 0, 3);
    }
    __syncthreads();

    // ---- S = Q * K^T for two 16-key tiles ----
    v8f s0 = {}, s1 = {};
#pragma unroll
    for (int kc = 0; kc < 8; ++kc) {
      v16h kb0 = load_Bt_frag(&Ksh[0][kc * 32], C_ + 8);
      v16h kb1 = load_Bt_frag(&Ksh[16][kc * 32], C_ + 8);
      s0 = wmma_f32(qf[kc], kb0, s0);
      s1 = wmma_f32(qf[kc], kb1, s1);
    }
    // ---- online softmax (row stats via width-16 lane reductions) ----
    float scale_old[8];
#pragma unroll
    for (int r = 0; r < 8; ++r) {
      float v0 = s0[r], v1 = s1[r];
      float mx = fmaxf(v0, v1);
#pragma unroll
      for (int off = 1; off < 16; off <<= 1)
        mx = fmaxf(mx, __shfl_xor(mx, off, 16));
      float mnew = fmaxf(mrow[r], mx);
      float p0 = __expf(v0 - mnew), p1 = __expf(v1 - mnew);
      float rs = p0 + p1;
#pragma unroll
      for (int off = 1; off < 16; off <<= 1)
        rs += __shfl_xor(rs, off, 16);
      scale_old[r] = __expf(mrow[r] - mnew);
      lrow[r] = lrow[r] * scale_old[r] + rs;
      mrow[r] = mnew;
      s0[r] = p0; s1[r] = p1;                            // reuse as P
    }
#pragma unroll
    for (int ct = 0; ct < 16; ++ct)
#pragma unroll
      for (int r = 0; r < 8; ++r) oacc[ct][r] *= scale_old[r];
    // ---- C/D-layout P -> LDS -> A-layout fragment (same-wave DS in order) ----
#pragma unroll
    for (int r = 0; r < 8; ++r) {
      Pl[wave][r + rowOff][colB]      = (f16)s0[r];
      Pl[wave][r + rowOff][16 + colB] = (f16)s1[r];
    }
    v16h pa = load_A_frag(&Pl[wave][0][0], 32);
    // ---- O += P * V (V^T tile gives contiguous per-lane reads) ----
#pragma unroll
    for (int ct = 0; ct < 16; ++ct) {
      v16h vb = load_Bt_frag(&Vsh[ct * 16][0], 40);
      oacc[ct] = wmma_f32(pa, vb, oacc[ct]);
    }
    __syncthreads();
  }
  float inv[8];
#pragma unroll
  for (int r = 0; r < 8; ++r) inv[r] = 1.0f / lrow[r];
  f16* Ob = O + (size_t)b * N_ * C_;
#pragma unroll
  for (int ct = 0; ct < 16; ++ct)
#pragma unroll
    for (int r = 0; r < 8; ++r)
      Ob[(size_t)(gq + r + rowOff) * C_ + ct * 16 + colB] = (f16)(oacc[ct][r] * inv[r]);
}

// ---------------- Output projection + bias + residual ------------------------
// out[b][o][n] = sum_c Wp[o][c]*O[b][n][c] + bp[o] + x[b][o][n]
__global__ void proj_out_k(const f16* __restrict__ O, const f16* __restrict__ Wp16,
                           const float* __restrict__ bp, const float* __restrict__ x,
                           float* __restrict__ out) {
  int wave = threadIdx.x >> 5, lane = threadIdx.x & 31;
  int gwave = blockIdx.x * 8 + wave;                     // 4096 waves
  int tile = gwave & 511;
  int b    = gwave >> 9;
  int oT = (tile & 7) * 32;
  int nT = (tile >> 3) * 32;
  const f16* Ob = O + (size_t)b * N_ * C_;
  v8f acc[2][2] = {};
  for (int k = 0; k < C_; k += 32) {
    v16h a0 = load_A_frag(Wp16 + (size_t)oT * C_ + k, C_);
    v16h a1 = load_A_frag(Wp16 + (size_t)(oT + 16) * C_ + k, C_);
    v16h b0 = load_Bt_frag(Ob + (size_t)nT * C_ + k, C_);
    v16h b1 = load_Bt_frag(Ob + (size_t)(nT + 16) * C_ + k, C_);
    acc[0][0] = wmma_f32(a0, b0, acc[0][0]);
    acc[0][1] = wmma_f32(a0, b1, acc[0][1]);
    acc[1][0] = wmma_f32(a1, b0, acc[1][0]);
    acc[1][1] = wmma_f32(a1, b1, acc[1][1]);
  }
  int rowOff = (lane >> 4) << 3, colB = lane & 15;
#pragma unroll
  for (int mi = 0; mi < 2; ++mi)
#pragma unroll
    for (int ni = 0; ni < 2; ++ni)
#pragma unroll
      for (int r = 0; r < 8; ++r) {
        int o = oT + mi * 16 + r + rowOff;
        int n = nT + ni * 16 + colB;
        size_t idx = (size_t)b * C_ * N_ + (size_t)o * N_ + n;
        out[idx] = acc[mi][ni][r] + bp[o] + x[idx];
      }
}

extern "C" void kernel_launch(void* const* d_in, const int* in_sizes, int n_in,
                              void* d_out, int out_size, void* d_ws, size_t ws_size,
                              hipStream_t stream) {
  const float* x     = (const float*)d_in[0];
  const float* gamma = (const float*)d_in[1];
  const float* beta  = (const float*)d_in[2];
  const float* Wq    = (const float*)d_in[3];
  const float* bq    = (const float*)d_in[4];
  const float* Wk    = (const float*)d_in[5];
  const float* bk    = (const float*)d_in[6];
  const float* Wv    = (const float*)d_in[7];
  const float* bv    = (const float*)d_in[8];
  const float* Wp    = (const float*)d_in[9];
  const float* bp    = (const float*)d_in[10];
  float* out = (float*)d_out;

  char* ws = (char*)d_ws;
  size_t off = 0;
  float* s = (float*)(ws + off); off += C_ * sizeof(float);
  float* t = (float*)(ws + off); off += C_ * sizeof(float);
  off = (off + 255) & ~(size_t)255;
  f16* W3   = (f16*)(ws + off); off += (size_t)3 * C_ * C_ * sizeof(f16);
  f16* Wp16 = (f16*)(ws + off); off += (size_t)C_ * C_ * sizeof(f16);
  f16* H    = (f16*)(ws + off); off += (size_t)B_ * C_ * N_ * sizeof(f16);
  f16* Qm   = (f16*)(ws + off); off += (size_t)B_ * N_ * C_ * sizeof(f16);
  f16* Km   = (f16*)(ws + off); off += (size_t)B_ * N_ * C_ * sizeof(f16);
  f16* Vm   = (f16*)(ws + off); off += (size_t)B_ * N_ * C_ * sizeof(f16);
  f16* Om   = (f16*)(ws + off); off += (size_t)B_ * N_ * C_ * sizeof(f16);

  dim3 blk(256);
  bn_stats_k<<<C_, blk, 0, stream>>>(x, gamma, beta, s, t);
  bn_apply_k<<<(B_ * C_ * N_) / 256, blk, 0, stream>>>(x, s, t, H);
  cvt_w_k<<<(C_ * C_) / 256, blk, 0, stream>>>(Wq, Wk, Wv, Wp, W3, Wp16);
  proj_qkv_k<<<1536, blk, 0, stream>>>(H, W3, bq, bk, bv, Qm, Km, Vm);
  attn_k<<<128, blk, 0, stream>>>(Qm, Km, Vm, Om);
  proj_out_k<<<512, blk, 0, stream>>>(Om, Wp16, bp, x, out);
}
